// CMSSelfModifyingEncoder_84164179132976
// MI455X (gfx1250) — matrix operand order repaired and checked by reading, not compiled, not run on hardware
//
#include <hip/hip_runtime.h>
#include <hip/hip_bf16.h>
#include <math.h>

typedef __bf16 bf16;
typedef __attribute__((ext_vector_type(16))) __bf16 v16bf;
typedef __attribute__((ext_vector_type(8)))  float  v8f;

#define DEV __device__ __forceinline__

constexpr int   BATCH   = 32;
constexpr int   C_IN    = 256;
constexpr int   HID     = 512;
constexpr int   BOTC    = 64;
constexpr int   HW      = 64 * 64;            // 4096
constexpr long long NPIX = (long long)BATCH * HW;   // 131072
constexpr int   GROUPS  = 8;
constexpr int   GCH     = HID / GROUPS;       // 64
constexpr float GN_EPS   = 1e-5f;
constexpr float INNER_LR = 0.005f;
constexpr float CONSIST_W = 0.1f;

DEV float geluf(float x) { return 0.5f * x * (1.f + erff(x * 0.7071067811865476f)); }
DEV float dgeluf(float x) {
    float cdf = 0.5f * (1.f + erff(x * 0.7071067811865476f));
    float pdf = 0.3989422804014327f * __expf(-0.5f * x * x);
    return cdf + x * pdf;
}

// CDNA5 async global->LDS copy (ASYNCcnt-tracked), 16B per lane.
DEV void async_ld_b128(const bf16* gaddr, unsigned lds_off) {
    asm volatile("global_load_async_to_lds_b128 %0, %1, off"
                 :: "v"(lds_off), "v"(gaddr) : "memory");
}
DEV unsigned lds_off_of(const void* p) {
    // generic LDS pointer = {aperture, 32-bit LDS offset}; low 32 bits are the DS address
    return (unsigned)(unsigned long long)p;
}

// ---------------------------------------------------------------------------
// Generic WMMA GEMM: Out[m,j] = sum_k A[m,k] * W[j,k]
// A: bf16 [M,K] row-major, W: bf16 [J,K] row-major, Out: f32 [M,J].
// Block = 256 threads = 8 waves; tile 64(M) x 64(J), K-step 32.
// Double-buffered LDS filled by async global->LDS loads (prefetch 1 tile ahead).
// ---------------------------------------------------------------------------
__global__ __launch_bounds__(256) void k_gemm(float* __restrict__ out,
        const bf16* __restrict__ A, const bf16* __restrict__ W,
        int M, int K, int J)
{
    __shared__ __attribute__((aligned(16))) bf16 As[2][64][32];
    __shared__ __attribute__((aligned(16))) bf16 Ws[2][64][32];
    const int tid  = threadIdx.x;
    const int lane = tid & 31;
    const int wv   = tid >> 5;
    const int wm   = wv & 3;
    const int wj   = wv >> 2;
    const int m0   = blockIdx.x * 64;
    const int j0   = blockIdx.y * 64;
    const int lr   = tid >> 2;           // 0..63 row for cooperative fill
    const int lc   = (tid & 3) * 8;      // 8 bf16 (16B) per thread
    v8f c0 = {}; v8f c1 = {};

    const int nk = K / 32;
    // prologue: async-issue K-tile 0 into buffer 0
    async_ld_b128(A + (size_t)(m0 + lr) * K + lc, lds_off_of(&As[0][lr][lc]));
    async_ld_b128(W + (size_t)(j0 + lr) * K + lc, lds_off_of(&Ws[0][lr][lc]));

    for (int it = 0; it < nk; ++it) {
        const int buf = it & 1;
        if (it + 1 < nk) {
            int k0 = (it + 1) * 32;
            async_ld_b128(A + (size_t)(m0 + lr) * K + k0 + lc, lds_off_of(&As[buf ^ 1][lr][lc]));
            async_ld_b128(W + (size_t)(j0 + lr) * K + k0 + lc, lds_off_of(&Ws[buf ^ 1][lr][lc]));
            asm volatile("s_wait_asynccnt 0x2" ::: "memory");  // current tile landed
        } else {
            asm volatile("s_wait_asynccnt 0x0" ::: "memory");
        }
        __syncthreads();
        // A fragment: 16x32 bf16, contiguous k = 16*(lane>>4) + element
        v16bf a;
        {
            int m    = wm * 16 + (lane & 15);
            int half = lane >> 4;
#pragma unroll
            for (int jj = 0; jj < 8; ++jj) {
                int kk = half * 16 + jj * 2;
                a[2 * jj]     = As[buf][m][kk];
                a[2 * jj + 1] = As[buf][m][kk + 1];
            }
        }
        // B fragments: B[k][n] = W[j0+n][k]
        v16bf b0, b1;
        {
            int khalf = (lane >> 4) * 16;
            int n     = wj * 32 + (lane & 15);
#pragma unroll
            for (int i = 0; i < 16; ++i) {
                b0[i] = Ws[buf][n][khalf + i];
                b1[i] = Ws[buf][n + 16][khalf + i];
            }
        }
        c0 = __builtin_amdgcn_wmma_f32_16x16x32_bf16(false, a, false, b0, (short)0, c0, false, false);
        c1 = __builtin_amdgcn_wmma_f32_16x16x32_bf16(false, a, false, b1, (short)0, c1, false, false);
        __syncthreads();   // protect buffer reuse by next async issue
    }
    // C layout: VGPR i -> row = i + 8*(lane>>4), col = lane&15
    int nn    = lane & 15;
    int rbase = m0 + wm * 16 + (lane >> 4) * 8;
#pragma unroll
    for (int i = 0; i < 8; ++i) {
        size_t row = (size_t)(rbase + i);
        out[row * J + j0 + wj * 32 + nn]      = c0[i];
        out[row * J + j0 + wj * 32 + 16 + nn] = c1[i];
    }
}

// ---------------------------------------------------------------------------
// Split-K reduction GEMM: Out[m,j] += sum_n A1[n,m] * A2[n,j]  (fp32 atomics)
// A1: bf16 [N,M1], A2: bf16 [N,M2], Out: f32 [M1,M2].
// blockIdx.x enumerates (M1/64)*(M2/64) tiles, blockIdx.y = N chunk.
// Same async double-buffered LDS pipeline.
// ---------------------------------------------------------------------------
__global__ __launch_bounds__(256) void k_gemm_nt(float* __restrict__ out,
        const bf16* __restrict__ A1, const bf16* __restrict__ A2,
        int Ntot, int M1, int M2, int nchunk)
{
    __shared__ __attribute__((aligned(16))) bf16 T1[2][32][64];
    __shared__ __attribute__((aligned(16))) bf16 T2[2][32][64];
    const int tid  = threadIdx.x;
    const int lane = tid & 31;
    const int wv   = tid >> 5;
    const int wm   = wv & 3;
    const int wj   = wv >> 2;
    const int tilesM = M1 / 64;
    const int m0 = (blockIdx.x % tilesM) * 64;
    const int j0 = (blockIdx.x / tilesM) * 64;
    long long nbeg = (long long)blockIdx.y * nchunk;
    long long nend = nbeg + nchunk; if (nend > Ntot) nend = Ntot;
    const int lr  = tid >> 3;            // 0..31
    const int lc  = (tid & 7) * 8;       // 8 bf16 per thread
    v8f c0 = {}; v8f c1 = {};

    const int nit = (int)((nend - nbeg) / 32);
    async_ld_b128(A1 + (nbeg + lr) * M1 + m0 + lc, lds_off_of(&T1[0][lr][lc]));
    async_ld_b128(A2 + (nbeg + lr) * M2 + j0 + lc, lds_off_of(&T2[0][lr][lc]));

    for (int it = 0; it < nit; ++it) {
        const int buf = it & 1;
        if (it + 1 < nit) {
            long long n0 = nbeg + (long long)(it + 1) * 32;
            async_ld_b128(A1 + (n0 + lr) * M1 + m0 + lc, lds_off_of(&T1[buf ^ 1][lr][lc]));
            async_ld_b128(A2 + (n0 + lr) * M2 + j0 + lc, lds_off_of(&T2[buf ^ 1][lr][lc]));
            asm volatile("s_wait_asynccnt 0x2" ::: "memory");
        } else {
            asm volatile("s_wait_asynccnt 0x0" ::: "memory");
        }
        __syncthreads();
        v16bf a;
        {
            int m    = wm * 16 + (lane & 15);
            int half = lane >> 4;
#pragma unroll
            for (int jj = 0; jj < 8; ++jj) {
                int kk = half * 16 + jj * 2;
                a[2 * jj]     = T1[buf][kk][m];
                a[2 * jj + 1] = T1[buf][kk + 1][m];
            }
        }
        v16bf b0, b1;
        {
            int khalf = (lane >> 4) * 16;
            int n     = wj * 32 + (lane & 15);
#pragma unroll
            for (int i = 0; i < 16; ++i) {
                b0[i] = T2[buf][khalf + i][n];
                b1[i] = T2[buf][khalf + i][n + 16];
            }
        }
        c0 = __builtin_amdgcn_wmma_f32_16x16x32_bf16(false, a, false, b0, (short)0, c0, false, false);
        c1 = __builtin_amdgcn_wmma_f32_16x16x32_bf16(false, a, false, b1, (short)0, c1, false, false);
        __syncthreads();
    }
    int nn    = lane & 15;
    int rbase = m0 + wm * 16 + (lane >> 4) * 8;
#pragma unroll
    for (int i = 0; i < 8; ++i) {
        size_t row = (size_t)(rbase + i);
        atomicAdd(&out[row * M2 + j0 + wj * 32 + nn],      c0[i]);
        atomicAdd(&out[row * M2 + j0 + wj * 32 + 16 + nn], c1[i]);
    }
}

// ---------------------------------------------------------------------------
// Elementwise / reduction kernels
// ---------------------------------------------------------------------------
__global__ void k_zero(float* p, long long n) {
    long long i = (long long)blockIdx.x * blockDim.x + threadIdx.x;
    long long st = (long long)gridDim.x * blockDim.x;
    for (; i < n; i += st) p[i] = 0.f;
}

__global__ void k_cast(bf16* o, const float* w, long long n) {
    long long i = (long long)blockIdx.x * blockDim.x + threadIdx.x;
    long long st = (long long)gridDim.x * blockDim.x;
    for (; i < n; i += st) o[i] = (bf16)w[i];
}

__global__ void k_castT(bf16* o, const float* w, int R, int Cn) {
    long long n = (long long)R * Cn;
    long long i = (long long)blockIdx.x * blockDim.x + threadIdx.x;
    long long st = (long long)gridDim.x * blockDim.x;
    for (; i < n; i += st) {
        int r = (int)(i / Cn), c = (int)(i % Cn);
        o[(long long)c * R + r] = (bf16)w[i];
    }
}

__global__ void k_update_cast(bf16* o, const float* w, const float* g, long long n) {
    long long i = (long long)blockIdx.x * blockDim.x + threadIdx.x;
    long long st = (long long)gridDim.x * blockDim.x;
    for (; i < n; i += st) o[i] = (bf16)(w[i] - INNER_LR * g[i]);
}

__global__ void k_update_f32(float* o, const float* w, const float* g, long long n) {
    long long i = (long long)blockIdx.x * blockDim.x + threadIdx.x;
    long long st = (long long)gridDim.x * blockDim.x;
    for (; i < n; i += st) o[i] = w[i] - INNER_LR * g[i];
}

// NCHW f32 -> [N,C] bf16 (tiled transpose)
__global__ __launch_bounds__(256) void k_x_to_nc(bf16* xb, const float* x) {
    __shared__ float tile[32][33];
    int b = blockIdx.z;
    int hw0 = blockIdx.x * 32;
    int c0 = blockIdx.y * 32;
    int tx = threadIdx.x & 31, ty = threadIdx.x >> 5;
#pragma unroll
    for (int yy = 0; yy < 32; yy += 8) {
        int c = c0 + ty + yy;
        tile[ty + yy][tx] = x[(long long)b * 1048576 + (long long)c * 4096 + hw0 + tx];
    }
    __syncthreads();
#pragma unroll
    for (int yy = 0; yy < 32; yy += 8) {
        int hw = hw0 + ty + yy;
        xb[((long long)b * 4096 + hw) * 256 + c0 + tx] = (bf16)tile[tx][ty + yy];
    }
}

__global__ __launch_bounds__(256) void k_masksum(const int* mask, float* msum) {
    __shared__ float sh[256];
    float loc = 0.f;
    long long i = (long long)blockIdx.x * blockDim.x + threadIdx.x;
    long long st = (long long)gridDim.x * blockDim.x;
    for (; i < NPIX; i += st) loc += (mask[i] == 0) ? 1.f : 0.f;
    sh[threadIdx.x] = loc; __syncthreads();
    for (int o = 128; o > 0; o >>= 1) {
        if (threadIdx.x < o) sh[threadIdx.x] += sh[threadIdx.x + o];
        __syncthreads();
    }
    if (threadIdx.x == 0) atomicAdd(msum, sh[0]);
}

// per-(b,group) mean / biased var of h [N,512]
__global__ __launch_bounds__(256) void k_group_stats(const float* h, float* gm, float* gv) {
    int bg = blockIdx.x;             // 0..255
    int b = bg >> 3, g = bg & 7;
    const long long base = ((long long)b * 4096) * 512 + g * 64;
    float s = 0.f, s2 = 0.f;
    for (int idx = threadIdx.x; idx < GCH * HW; idx += 256) {
        int hw = idx >> 6, kk = idx & 63;
        float v = h[base + (long long)hw * 512 + kk];
        s += v; s2 += v * v;
    }
    __shared__ float ss[256], sq[256];
    ss[threadIdx.x] = s; sq[threadIdx.x] = s2; __syncthreads();
    for (int o = 128; o > 0; o >>= 1) {
        if (threadIdx.x < o) { ss[threadIdx.x] += ss[threadIdx.x + o]; sq[threadIdx.x] += sq[threadIdx.x + o]; }
        __syncthreads();
    }
    if (threadIdx.x == 0) {
        const float inv = 1.f / (float)(GCH * HW);
        float m = ss[0] * inv;
        gm[bg] = m;
        gv[bg] = sq[0] * inv - m * m;
    }
}

__global__ void k_gn_gelu(const float* h, const float* gm, const float* gv,
                          const float* gnw, const float* gnb, bf16* hn_out, bf16* g_out) {
    long long tot = NPIX * HID;
    long long i = (long long)blockIdx.x * blockDim.x + threadIdx.x;
    long long st = (long long)gridDim.x * blockDim.x;
    for (; i < tot; i += st) {
        int k = (int)(i & 511);
        long long n = i >> 9;
        int bg = ((int)(n >> 12)) * 8 + (k >> 6);
        float rinv = rsqrtf(gv[bg] + GN_EPS);
        float hn = (h[i] - gm[bg]) * rinv;
        float gn = hn * gnw[k] + gnb[k];
        hn_out[i] = (bf16)hn;
        g_out[i]  = (bf16)geluf(gn);
    }
}

// z = mod + x ; zmask = z*mask (bf16) ; per-channel sum/sumsq atomics
__global__ __launch_bounds__(256) void k_z_prep(const float* mod, const float* x, const int* mask,
        bf16* zm, float* ch_sum, float* ch_sq) {
    int c = threadIdx.x;
    long long n0 = (long long)blockIdx.x * 64;
    float s = 0.f, s2 = 0.f;
    for (int r = 0; r < 64; ++r) {
        long long n = n0 + r;
        int b = (int)(n >> 12), hw = (int)(n & 4095);
        float z = mod[n * 256 + c] + x[(long long)b * 1048576 + (long long)c * 4096 + hw];
        s += z; s2 += z * z;
        zm[n * 256 + c] = (bf16)(z * (float)mask[n]);
    }
    atomicAdd(&ch_sum[c], s); atomicAdd(&ch_sq[c], s2);
}

__global__ void k_gelu_hr(const float* hp, bf16* hr) {
    long long tot = NPIX * BOTC;
    long long i = (long long)blockIdx.x * blockDim.x + threadIdx.x;
    long long st = (long long)gridDim.x * blockDim.x;
    for (; i < tot; i += st) hr[i] = (bf16)geluf(hp[i]);
}

// drec (bf16) and dz init (= -drec + consistency term)
__global__ __launch_bounds__(256) void k_drec_dz(const float* rec, const float* mod, const float* x,
        const int* mask, const float* ch_sum, const float* ch_sq, const float* msum,
        bf16* drec, float* dz) {
    int c = threadIdx.x;
    const float NB = (float)NPIX;
    float cm = ch_sum[c] / NB;
    float cv = (ch_sq[c] - NB * cm * cm) / (NB - 1.f);
    float D = msum[0] * 256.f + 1e-8f;
    const float czk = 1.f / (1.f + 1e-8f);
    float cterm = CONSIST_W * 2.f * cm / (256.f * NB);
    float cfac  = CONSIST_W * (2.f * (cv * czk - 1.f) * czk / 256.f) * (2.f / (NB - 1.f));
    long long n0 = (long long)blockIdx.x * 64;
    for (int r = 0; r < 64; ++r) {
        long long n = n0 + r;
        int b = (int)(n >> 12), hw = (int)(n & 4095);
        float z = mod[n * 256 + c] + x[(long long)b * 1048576 + (long long)c * 4096 + hw];
        float inv = 1.f - (float)mask[n];
        float dr = 2.f * (rec[n * 256 + c] - z) * inv / D;
        drec[n * 256 + c] = (bf16)dr;
        dz[n * 256 + c] = -dr + cterm + cfac * (z - cm);
    }
}

__global__ void k_dhrpre(const float* hrgrad, const float* hrpre, bf16* o) {
    long long tot = NPIX * BOTC;
    long long i = (long long)blockIdx.x * blockDim.x + threadIdx.x;
    long long st = (long long)gridDim.x * blockDim.x;
    for (; i < tot; i += st) o[i] = (bf16)(hrgrad[i] * dgeluf(hrpre[i]));
}

__global__ void k_dz_add(float* dz, const float* dmask, const int* mask, bf16* dzb) {
    long long tot = NPIX * 256;
    long long i = (long long)blockIdx.x * blockDim.x + threadIdx.x;
    long long st = (long long)gridDim.x * blockDim.x;
    for (; i < tot; i += st) {
        long long n = i >> 8;
        float v = dz[i] + (float)mask[n] * dmask[i];
        dz[i] = v;
        dzb[i] = (bf16)v;
    }
}

// dg -> dgn -> dhn (in place); accumulate dgnw/dgnb and group sums S1/S2
__global__ __launch_bounds__(256) void k_dgn(float* dg_dhn, const bf16* hn,
        const float* gnw, const float* gnb, float* dgnw, float* dgnb, float* S1, float* S2) {
    long long n0 = (long long)blockIdx.x * 64;
    int b = (int)(n0 >> 12);
    int t = threadIdx.x;
    __shared__ float sS[16];
    if (t < 16) sS[t] = 0.f;
    __syncthreads();
    int c0 = t, c1 = t + 256;
    float w0 = gnw[c0], b0v = gnb[c0], w1v = gnw[c1], b1v = gnb[c1];
    float aW0 = 0, aB0 = 0, a10 = 0, a20 = 0, aW1 = 0, aB1 = 0, a11 = 0, a21 = 0;
    for (int r = 0; r < 64; ++r) {
        long long base = (n0 + r) * 512;
        float h0 = (float)hn[base + c0];
        float dgn0 = dg_dhn[base + c0] * dgeluf(h0 * w0 + b0v);
        float dhn0 = dgn0 * w0;
        dg_dhn[base + c0] = dhn0;
        aW0 += dgn0 * h0; aB0 += dgn0; a10 += dhn0; a20 += dhn0 * h0;
        float h1 = (float)hn[base + c1];
        float dgn1 = dg_dhn[base + c1] * dgeluf(h1 * w1v + b1v);
        float dhn1 = dgn1 * w1v;
        dg_dhn[base + c1] = dhn1;
        aW1 += dgn1 * h1; aB1 += dgn1; a11 += dhn1; a21 += dhn1 * h1;
    }
    atomicAdd(&dgnw[c0], aW0); atomicAdd(&dgnb[c0], aB0);
    atomicAdd(&dgnw[c1], aW1); atomicAdd(&dgnb[c1], aB1);
    int g0 = c0 >> 6, g1 = c1 >> 6;
    atomicAdd(&sS[g0], a10); atomicAdd(&sS[8 + g0], a20);
    atomicAdd(&sS[g1], a11); atomicAdd(&sS[8 + g1], a21);
    __syncthreads();
    if (t < 8) { atomicAdd(&S1[b * 8 + t], sS[t]); atomicAdd(&S2[b * 8 + t], sS[8 + t]); }
}

__global__ void k_gn_bwd(const float* dhn, const bf16* hn, const float* gv,
                         const float* S1, const float* S2, bf16* dhp) {
    long long tot = NPIX * HID;
    const float invNg = 1.f / (float)(GCH * HW);
    long long i = (long long)blockIdx.x * blockDim.x + threadIdx.x;
    long long st = (long long)gridDim.x * blockDim.x;
    for (; i < tot; i += st) {
        int k = (int)(i & 511);
        long long n = i >> 9;
        int bg = ((int)(n >> 12)) * 8 + (k >> 6);
        float rinv = rsqrtf(gv[bg] + GN_EPS);
        float v = rinv * (dhn[i] - S1[bg] * invNg - (float)hn[i] * (S2[bg] * invNg));
        dhp[i] = (bf16)v;
    }
}

__global__ __launch_bounds__(256) void k_pool(const float* x, float* pooled) {
    int bc = blockIdx.x;                   // b*256 + c
    int b = bc >> 8, c = bc & 255;
    const float* p = x + (long long)b * 1048576 + (long long)c * 4096;
    float s = 0.f;
    for (int i = threadIdx.x; i < 4096; i += 256) s += p[i];
    __shared__ float sh[256];
    sh[threadIdx.x] = s; __syncthreads();
    for (int o = 128; o > 0; o >>= 1) {
        if (threadIdx.x < o) sh[threadIdx.x] += sh[threadIdx.x + o];
        __syncthreads();
    }
    if (threadIdx.x == 0) pooled[bc] = sh[0] * (1.f / 4096.f);
}

__global__ void k_gate(const float* pooled, const float* gw1, const float* gb1,
                       const float* gw2, const float* gb2, float* gate) {
    int b = blockIdx.x;
    int j = threadIdx.x;                   // 64 threads
    __shared__ float sh[64];
    float s = gb1[j];
    for (int c = 0; c < 256; ++c) s += pooled[b * 256 + c] * gw1[j * 256 + c];
    sh[j] = geluf(s);
    __syncthreads();
    if (j == 0) {
        float t = gb2[0];
        for (int q = 0; q < 64; ++q) t += sh[q] * gw2[q];
        gate[b] = 1.f / (1.f + __expf(-t));
    }
}

__global__ void k_final(float* out, const float* x, const float* mod2,
                        const float* gate, const float* rs) {
    long long tot = NPIX * 256;
    long long i = (long long)blockIdx.x * blockDim.x + threadIdx.x;
    long long st = (long long)gridDim.x * blockDim.x;
    for (; i < tot; i += st) {
        long long n = i >> 8;
        int c = (int)(i & 255);
        int b = (int)(n >> 12), hw = (int)(n & 4095);
        long long a = (long long)b * 1048576 + (long long)c * 4096 + hw;
        out[a] = x[a] + rs[0] * gate[b] * mod2[i];
    }
}

// ---------------------------------------------------------------------------
extern "C" void kernel_launch(void* const* d_in, const int* in_sizes, int n_in,
                              void* d_out, int out_size, void* d_ws, size_t ws_size,
                              hipStream_t stream) {
    (void)in_sizes; (void)n_in; (void)out_size; (void)ws_size;
    const float* x   = (const float*)d_in[0];
    const int*   msk = (const int*)d_in[1];
    const float* w1  = (const float*)d_in[2];
    const float* gnw = (const float*)d_in[3];
    const float* gnb = (const float*)d_in[4];
    const float* w2  = (const float*)d_in[5];
    const float* rc1 = (const float*)d_in[6];
    const float* rc2 = (const float*)d_in[7];
    const float* gw1 = (const float*)d_in[8];
    const float* gb1 = (const float*)d_in[9];
    const float* gw2 = (const float*)d_in[10];
    const float* gb2 = (const float*)d_in[11];
    const float* rs  = (const float*)d_in[12];
    float* out = (float*)d_out;

    char* wsb = (char*)d_ws;
    size_t off = 0;
    auto alloc = [&](size_t bytes) -> void* {
        void* p = wsb + off;
        off += (bytes + 255) & ~(size_t)255;
        return p;
    };
    const size_t N = 131072;
    bf16*  XB  = (bf16*)alloc(N * 256 * 2);     // x as [N,C] bf16
    float* B1  = (float*)alloc(N * 512 * 4);    // hpre / dg / dhn / hpre2
    bf16*  HN  = (bf16*)alloc(N * 512 * 2);     // normalized hidden
    bf16*  GA  = (bf16*)alloc(N * 512 * 2);     // gelu hidden (GEMM operand)
    float* MOD = (float*)alloc(N * 256 * 4);    // modifier out / mod2
    float* DZ  = (float*)alloc(N * 256 * 4);
    bf16*  ZM  = (bf16*)alloc(N * 256 * 2);     // masked z
    float* HRP = (float*)alloc(N * 64 * 4);     // reconstructor pre-act
    bf16*  HR  = (bf16*)alloc(N * 64 * 2);      // hr, later dhrpre
    float* JF  = (float*)alloc(N * 512 * 2);    // rec / hrgrad / dmasked, then dhpre bf16
    bf16*  JB  = (bf16*)JF;
    bf16*  DZB = (bf16*)alloc(N * 256 * 2);     // drec, later dz bf16
    // accumulator region (zeroed each call)
    float* ACC  = (float*)alloc((2112 + 262144) * sizeof(float));
    float* CHS  = ACC;          float* CHQ  = ACC + 256;
    float* S1   = ACC + 512;    float* S2   = ACC + 768;
    float* DGNW = ACC + 1024;   float* DGNB = ACC + 1536;
    float* MSUM = ACC + 2048;
    float* DW1  = ACC + 2112;   float* DW2  = DW1 + 131072;
    const long long accN = 2112 + 262144;
    // small non-accumulated
    float* GM  = (float*)alloc(256 * 4);  float* GV  = (float*)alloc(256 * 4);
    float* GM2 = (float*)alloc(256 * 4);  float* GV2 = (float*)alloc(256 * 4);
    bf16* W1B  = (bf16*)alloc(131072 * 2);
    bf16* W2B  = (bf16*)alloc(131072 * 2);
    bf16* W2T  = (bf16*)alloc(131072 * 2);
    bf16* RC1B = (bf16*)alloc(16384 * 2);
    bf16* RC1T = (bf16*)alloc(16384 * 2);
    bf16* RC2B = (bf16*)alloc(16384 * 2);
    bf16* RC2T = (bf16*)alloc(16384 * 2);
    bf16* W1U  = (bf16*)alloc(131072 * 2);
    bf16* W2U  = (bf16*)alloc(131072 * 2);
    float* GNWU = (float*)alloc(512 * 4);
    float* GNBU = (float*)alloc(512 * 4);
    float* POOL = (float*)alloc(8192 * 4);
    float* GATE = (float*)alloc(32 * 4);

    const int M = (int)N;
    // ---- setup
    k_zero<<<256, 256, 0, stream>>>(ACC, accN);
    k_cast<<<64, 256, 0, stream>>>(W1B, w1, 131072);
    k_cast<<<64, 256, 0, stream>>>(W2B, w2, 131072);
    k_cast<<<16, 256, 0, stream>>>(RC1B, rc1, 16384);
    k_cast<<<16, 256, 0, stream>>>(RC2B, rc2, 16384);
    k_castT<<<64, 256, 0, stream>>>(W2T, w2, 256, 512);
    k_castT<<<16, 256, 0, stream>>>(RC1T, rc1, 64, 256);
    k_castT<<<16, 256, 0, stream>>>(RC2T, rc2, 256, 64);
    k_x_to_nc<<<dim3(128, 8, 32), 256, 0, stream>>>(XB, x);
    k_masksum<<<256, 256, 0, stream>>>(msk, MSUM);
    // ---- inner-loss forward
    k_gemm<<<dim3(2048, 8), 256, 0, stream>>>(B1, XB, W1B, M, 256, 512);       // hpre
    k_group_stats<<<256, 256, 0, stream>>>(B1, GM, GV);
    k_gn_gelu<<<2048, 256, 0, stream>>>(B1, GM, GV, gnw, gnb, HN, GA);
    k_gemm<<<dim3(2048, 4), 256, 0, stream>>>(MOD, GA, W2B, M, 512, 256);      // mod
    k_z_prep<<<2048, 256, 0, stream>>>(MOD, x, msk, ZM, CHS, CHQ);
    k_gemm<<<dim3(2048, 1), 256, 0, stream>>>(HRP, ZM, RC1B, M, 256, 64);      // hr pre
    k_gelu_hr<<<2048, 256, 0, stream>>>(HRP, HR);
    k_gemm<<<dim3(2048, 4), 256, 0, stream>>>(JF, HR, RC2B, M, 64, 256);       // rec
    // ---- backward to modifier params
    k_drec_dz<<<2048, 256, 0, stream>>>(JF, MOD, x, msk, CHS, CHQ, MSUM, DZB, DZ);
    k_gemm<<<dim3(2048, 1), 256, 0, stream>>>(JF, DZB, RC2T, M, 256, 64);      // dhr
    k_dhrpre<<<2048, 256, 0, stream>>>(JF, HRP, HR);                           // dhrpre (bf16)
    k_gemm<<<dim3(2048, 4), 256, 0, stream>>>(JF, HR, RC1T, M, 64, 256);       // dmasked
    k_dz_add<<<2048, 256, 0, stream>>>(DZ, JF, msk, DZB);                      // dz, dzb
    k_gemm<<<dim3(2048, 8), 256, 0, stream>>>(B1, DZB, W2T, M, 256, 512);      // dg
    k_dgn<<<2048, 256, 0, stream>>>(B1, HN, gnw, gnb, DGNW, DGNB, S1, S2);     // dhn in place
    k_gn_bwd<<<2048, 256, 0, stream>>>(B1, HN, GV, S1, S2, JB);                // dhpre bf16
    k_gemm_nt<<<dim3(32, 16), 256, 0, stream>>>(DW2, DZB, GA, M, 256, 512, 8192);
    k_gemm_nt<<<dim3(32, 16), 256, 0, stream>>>(DW1, JB, XB, M, 512, 256, 8192);
    // ---- weight update (bufs = grads; p' = p - lr*grad)
    k_update_cast<<<64, 256, 0, stream>>>(W1U, w1, DW1, 131072);
    k_update_cast<<<64, 256, 0, stream>>>(W2U, w2, DW2, 131072);
    k_update_f32<<<4, 256, 0, stream>>>(GNWU, gnw, DGNW, 512);
    k_update_f32<<<4, 256, 0, stream>>>(GNBU, gnb, DGNB, 512);
    // ---- final modifier pass with fast weights
    k_gemm<<<dim3(2048, 8), 256, 0, stream>>>(B1, XB, W1U, M, 256, 512);
    k_group_stats<<<256, 256, 0, stream>>>(B1, GM2, GV2);
    k_gn_gelu<<<2048, 256, 0, stream>>>(B1, GM2, GV2, GNWU, GNBU, HN, GA);
    k_gemm<<<dim3(2048, 4), 256, 0, stream>>>(MOD, GA, W2U, M, 512, 256);      // mod2
    // ---- gate + residual
    k_pool<<<8192, 256, 0, stream>>>(x, POOL);
    k_gate<<<32, 64, 0, stream>>>(POOL, gw1, gb1, gw2, gb2, GATE);
    k_final<<<2048, 256, 0, stream>>>(out, x, MOD, GATE, rs);
}
